// H1SimplifiedPretrained_8770323219103
// MI455X (gfx1250) — compile-verified, matrix operand order checked by reading, not run on hardware
//
#include <hip/hip_runtime.h>
#include <hip/hip_bf16.h>
#include <math.h>

typedef _Float16 half_t;
typedef __attribute__((ext_vector_type(8)))  _Float16 v8h;
typedef __attribute__((ext_vector_type(16))) _Float16 v16h;
typedef __attribute__((ext_vector_type(8)))  float    v8f;

#define BDIM   1024
#define NCORP  200000
#define EDIM   128
#define HDIM   256
#define KTOP   32
#define ODIM   8
#define ROWS   32                 /* sim rows per block */
#define CHUNK  1024               /* sim cols per block */
#define NCHUNK 196                /* ceil(200000/1024) */
#define NCAND  (NCHUNK*KTOP)      /* 6272 */

#define NEG_INF (-__builtin_inff())

// ---------- WMMA fragment loader: 16-bit A/B layout (ISA 7.12.2) ----------
__device__ __forceinline__ v16h load_frag16(const half_t* p) {
  v8h lo = *(const v8h*)(p);
  v8h hi = *(const v8h*)(p + 16);
  return __builtin_shufflevector(lo, hi, 0,1,2,3,4,5,6,7,8,9,10,11,12,13,14,15);
}
__device__ __forceinline__ v8f wmma16(v16h a, v16h b, v8f c) {
  return __builtin_amdgcn_wmma_f32_16x16x32_f16(false, a, false, b, (short)0, c, false, false);
}

// ---------------- corpus row L2-normalize -> fp16 ----------------
__global__ void k_corpus_norm(const float* __restrict__ corpus, half_t* __restrict__ ch) {
  int lane = threadIdx.x & 31, wid = threadIdx.x >> 5;
  long row = (long)blockIdx.x * 8 + wid;
  if (row >= NCORP) return;
  const float4 v4 = *(const float4*)(corpus + row * EDIM + lane * 4);
  float s = v4.x*v4.x + v4.y*v4.y + v4.z*v4.z + v4.w*v4.w;
  #pragma unroll
  for (int o = 16; o; o >>= 1) s += __shfl_xor(s, o, 32);
  float n = sqrtf(s); n = fmaxf(n, 1e-12f);
  float sc = 1.0f / n;
  half_t* p = ch + row * EDIM + lane * 4;
  p[0] = (half_t)(v4.x * sc); p[1] = (half_t)(v4.y * sc);
  p[2] = (half_t)(v4.z * sc); p[3] = (half_t)(v4.w * sc);
}

// ---------------- generic fp32 linear: one block per row ----------------
__global__ void k_linear(const float* __restrict__ X, const float* __restrict__ W,
                         const float* __restrict__ bias, float* __restrict__ Y,
                         int K, int N, int ldy, int ycol0, int relu) {
  __shared__ float xs[768];
  int row = blockIdx.x, tid = threadIdx.x;
  for (int k = tid; k < K; k += 256) xs[k] = X[(size_t)row * K + k];
  __syncthreads();
  for (int n = tid; n < N; n += 256) {
    float acc = bias ? bias[n] : 0.0f;
    for (int k = 0; k < K; ++k) acc = fmaf(xs[k], W[(size_t)k * N + n], acc);
    if (relu) acc = fmaxf(acc, 0.0f);
    Y[(size_t)row * ldy + ycol0 + n] = acc;
  }
}

// ---------------- normalize embedding; emit fp32 output + fp16 query ----------------
__global__ void k_norm_query(const float* __restrict__ embRaw, float* __restrict__ outEmb,
                             half_t* __restrict__ queryH) {
  __shared__ float red[128];
  int b = blockIdx.x, t = threadIdx.x;
  float v = embRaw[b * EDIM + t];
  red[t] = v * v;
  __syncthreads();
  for (int s = 64; s; s >>= 1) { if (t < s) red[t] += red[t + s]; __syncthreads(); }
  float n = sqrtf(red[0]); n = fmaxf(n, 1e-12f);
  float q = v / n;
  outEmb[b * EDIM + t] = q;
  queryH[b * EDIM + t] = (half_t)q;
}

// ---------------- sim compute phase: explicit ping-pong double buffer ----------------
// TAIL=false: no bounds clamping at all (195 of 196 chunks).
template <bool TAIL>
__device__ __forceinline__ void sim_compute(const half_t* __restrict__ Q,
                                            const half_t* __restrict__ C,
                                            float* smem, int rowBase, int chunkBase,
                                            int wid, int lane) {
  const int m = lane & 15, koff = (lane >> 4) * 8;

  // two A-fragment sets: rows [rowBase,+16) and [rowBase+16,+32)
  v16h a0[4], a1[4];
  const half_t* q0 = Q + (size_t)(rowBase + m) * EDIM;
  const half_t* q1 = q0 + 16 * EDIM;
  #pragma unroll
  for (int i = 0; i < 4; ++i) {
    a0[i] = load_frag16(q0 + i * 32 + koff);
    a1[i] = load_frag16(q1 + i * 32 + koff);
  }

  const int colLocal0 = wid * 128;
  auto baddr = [&](int t) -> const half_t* {
    int gcol = chunkBase + colLocal0 + t * 16 + m;
    if (TAIL) gcol = gcol < NCORP ? gcol : (NCORP - 1);
    return C + (size_t)gcol * EDIM + koff;
  };
  auto compute = [&](v16h b0, v16h b1, v16h b2, v16h b3, int t) {
    v8f acc0 = {}, acc1 = {};
    acc0 = wmma16(a0[0], b0, acc0); acc1 = wmma16(a1[0], b0, acc1);
    acc0 = wmma16(a0[1], b1, acc0); acc1 = wmma16(a1[1], b1, acc1);
    acc0 = wmma16(a0[2], b2, acc0); acc1 = wmma16(a1[2], b2, acc1);
    acc0 = wmma16(a0[3], b3, acc0); acc1 = wmma16(a1[3], b3, acc1);
    // C/D layout: acc[r] -> row r + 8*(lane>=16), col = lane&15
    int colLocal = colLocal0 + t * 16 + (lane & 15);
    int rhi = (lane >> 4) * 8;
    bool ok = !TAIL || (chunkBase + colLocal) < NCORP;
    #pragma unroll
    for (int r = 0; r < 8; ++r) {
      smem[(r + rhi) * CHUNK + colLocal]      = ok ? acc0[r] : NEG_INF;
      smem[(16 + r + rhi) * CHUNK + colLocal] = ok ? acc1[r] : NEG_INF;
    }
  };

  // ping-pong buffers: no rotation copies at the loop back-edge
  v16h A0, A1, A2, A3, B0, B1, B2, B3;
  { const half_t* p = baddr(0);
    A0 = load_frag16(p); A1 = load_frag16(p + 32);
    A2 = load_frag16(p + 64); A3 = load_frag16(p + 96); }

  #pragma unroll 1
  for (int t = 0; t < 8; t += 2) {
    { const half_t* p = baddr(t + 1);            // load B while computing A
      B0 = load_frag16(p); B1 = load_frag16(p + 32);
      B2 = load_frag16(p + 64); B3 = load_frag16(p + 96); }
    compute(A0, A1, A2, A3, t);
    if (t + 2 < 8) {                             // load A while computing B
      const half_t* p = baddr(t + 2);
      A0 = load_frag16(p); A1 = load_frag16(p + 32);
      A2 = load_frag16(p + 64); A3 = load_frag16(p + 96);
    }
    compute(B0, B1, B2, B3, t + 1);
  }
}

// ---------------- sim (WMMA f16) + fused per-chunk top-32 ----------------
// grid (BDIM/32, NCHUNK), block 256, dynLDS = ROWS*CHUNK*4 bytes.
// launch_bounds(256,1): LDS already caps us at <=2 WG/WGP, so give the
// allocator the full VGPR budget (avoids scratch spills).
__global__ void __launch_bounds__(256, 1)
k_sim_topk(const half_t* __restrict__ Q, const half_t* __restrict__ C,
           float* __restrict__ cand_val, int* __restrict__ cand_idx) {
  extern __shared__ float smem[];           // [ROWS][CHUNK]
  const int tid = threadIdx.x;
  const int lane = tid & 31, wid = tid >> 5;
  const int rowBase = blockIdx.x * ROWS;
  const int chunkBase = blockIdx.y * CHUNK;

  if (blockIdx.y == NCHUNK - 1)
    sim_compute<true >(Q, C, smem, rowBase, chunkBase, wid, lane);
  else
    sim_compute<false>(Q, C, smem, rowBase, chunkBase, wid, lane);
  __syncthreads();

  // per-wave partial top-32: wave w owns rows 4w..4w+3 (loops uniform -> barriers safe)
  for (int rr = 0; rr < 4; ++rr) {
    int row = wid * 4 + rr;
    float* srow = smem + row * CHUNK;
    int grow = rowBase + row;
    float* cvp = cand_val + ((size_t)grow * NCHUNK + blockIdx.y) * KTOP;
    int*   cip = cand_idx + ((size_t)grow * NCHUNK + blockIdx.y) * KTOP;
    for (int sel = 0; sel < KTOP; ++sel) {
      float bv = NEG_INF; int bi = 0;
      for (int c = lane; c < CHUNK; c += 32) {
        float v = srow[c];
        if (v > bv) { bv = v; bi = c; }
      }
      #pragma unroll
      for (int o = 16; o; o >>= 1) {
        float ov = __shfl_xor(bv, o, 32);
        int   oi = __shfl_xor(bi, o, 32);
        if (ov > bv || (ov == bv && oi < bi)) { bv = ov; bi = oi; }
      }
      if (lane == 0) {
        cvp[sel] = bv;
        cip[sel] = chunkBase + bi;
        srow[bi] = NEG_INF;
      }
      __syncthreads();
    }
  }
}

// ---------------- merge per-chunk candidates -> final sorted top-32 ----------------
__global__ void k_merge_topk(const float* __restrict__ cand_val, const int* __restrict__ cand_idx,
                             float* __restrict__ scores, float* __restrict__ idxf,
                             int* __restrict__ idxi) {
  __shared__ float vals[NCAND];
  __shared__ int   idxs[NCAND];
  __shared__ float rv[8]; __shared__ int ri[8]; __shared__ int rp[8];
  int b = blockIdx.x, tid = threadIdx.x, lane = tid & 31, wid = tid >> 5;
  const float* cv = cand_val + (size_t)b * NCAND;
  const int*   ci = cand_idx + (size_t)b * NCAND;
  for (int i = tid; i < NCAND; i += 256) { vals[i] = cv[i]; idxs[i] = ci[i]; }
  __syncthreads();
  for (int sel = 0; sel < KTOP; ++sel) {
    float bv = NEG_INF; int bi = 0, bp = 0;
    for (int i = tid; i < NCAND; i += 256) {
      float v = vals[i];
      if (v > bv) { bv = v; bi = idxs[i]; bp = i; }
    }
    #pragma unroll
    for (int o = 16; o; o >>= 1) {
      float ov = __shfl_xor(bv, o, 32);
      int   oi = __shfl_xor(bi, o, 32);
      int   op = __shfl_xor(bp, o, 32);
      if (ov > bv) { bv = ov; bi = oi; bp = op; }
    }
    if (lane == 0) { rv[wid] = bv; ri[wid] = bi; rp[wid] = bp; }
    __syncthreads();
    if (tid == 0) {
      float best = rv[0]; int bbi = ri[0], bbp = rp[0];
      for (int i = 1; i < 8; ++i) if (rv[i] > best) { best = rv[i]; bbi = ri[i]; bbp = rp[i]; }
      scores[b * KTOP + sel] = best;
      idxf[b * KTOP + sel] = (float)bbi;
      idxi[b * KTOP + sel] = bbi;
      vals[bbp] = NEG_INF;
    }
    __syncthreads();
  }
}

// ---------------- gather retrieved rows via async memory->LDS DMA ----------------
__global__ void k_gather(const float* __restrict__ corpus, const int* __restrict__ topk,
                         half_t* __restrict__ flatH) {
  __shared__ float stage[KTOP * EDIM];      // 16 KB
  int b = blockIdx.x, tid = threadIdx.x;
  for (int i = tid; i < (KTOP * EDIM) / 4; i += 256) {   // 1024 x 16B segments
    int k  = i >> 5;               // retrieved row 0..31
    int e4 = (i & 31) * 4;         // float offset within row
    int row = topk[b * KTOP + k];
    const float* src = corpus + (size_t)row * EDIM + e4;
    unsigned lds = (unsigned)(size_t)(&stage[i * 4]);
    asm volatile("global_load_async_to_lds_b128 %0, %1, off"
                 :: "v"(lds), "v"(src) : "memory");
  }
  asm volatile("s_wait_asynccnt 0x0" ::: "memory");
  __syncthreads();
  for (int i = tid; i < KTOP * EDIM; i += 256)
    flatH[(size_t)b * (KTOP * EDIM) + i] = (half_t)stage[i];
}

// ---------------- W_r [4096,256] -> transposed fp16 [256,4096] ----------------
__global__ void k_wr_transpose(const float* __restrict__ Wr, half_t* __restrict__ Wrt) {
  size_t g = (size_t)blockIdx.x * 256 + threadIdx.x;   // over 4096*256
  int n = (int)(g & 255), k = (int)(g >> 8);
  Wrt[(size_t)n * 4096 + k] = (half_t)Wr[(size_t)k * 256 + n];
}

// ---------------- r_enc = flat @ W_r + b_r (WMMA), K=4096, double-buffered ----------------
// grid (64,16), block 32 (one wave per 16x16 output tile), writes combined[:,512:768]
__global__ void __launch_bounds__(32, 1)
k_renc_wmma(const half_t* __restrict__ A, const half_t* __restrict__ Bt,
            const float* __restrict__ bias, float* __restrict__ combined) {
  int lane = threadIdx.x & 31;
  int m = lane & 15, koff = (lane >> 4) * 8;
  int rowBase = blockIdx.x * 16, colBase = blockIdx.y * 16;
  const half_t* ar = A  + (size_t)(rowBase + m) * 4096 + koff;
  const half_t* br = Bt + (size_t)(colBase + m) * 4096 + koff;
  v8f acc = {};
  v16h af = load_frag16(ar), bf = load_frag16(br);
  #pragma unroll 4
  for (int kk = 0; kk < 4096; kk += 32) {
    v16h afn, bfn;
    if (kk + 32 < 4096) {
      __builtin_prefetch(ar + kk + 64, 0, 1);
      afn = load_frag16(ar + kk + 32);
      bfn = load_frag16(br + kk + 32);
    }
    acc = wmma16(af, bf, acc);
    if (kk + 32 < 4096) { af = afn; bf = bfn; }
  }
  int col = colBase + (lane & 15);
  int rhi = (lane >> 4) * 8;
  float bsv = bias[col];
  #pragma unroll
  for (int r = 0; r < 8; ++r)
    combined[(size_t)(rowBase + r + rhi) * 768 + 512 + col] = acc[r] + bsv;
}

// ---------------- a_in = [r_enc | outcome] ----------------
__global__ void k_build_ain(const float* __restrict__ combined, const float* __restrict__ outcome,
                            float* __restrict__ a_in) {
  int b = blockIdx.x, tid = threadIdx.x;
  for (int i = tid; i < 264; i += 256)
    a_in[(size_t)b * 264 + i] = (i < 256) ? combined[(size_t)b * 768 + 512 + i]
                                          : outcome[(size_t)b * ODIM + (i - 256)];
}

// ---------------- softmax over 32 (one wave per row) ----------------
__global__ void k_softmax32(const float* __restrict__ logits, float* __restrict__ out) {
  int lane = threadIdx.x & 31, wid = threadIdx.x >> 5;
  int b = blockIdx.x * 8 + wid;
  if (b >= BDIM) return;
  float v = logits[b * KTOP + lane];
  float mx = v;
  #pragma unroll
  for (int o = 16; o; o >>= 1) mx = fmaxf(mx, __shfl_xor(mx, o, 32));
  float e = __expf(v - mx);
  float s = e;
  #pragma unroll
  for (int o = 16; o; o >>= 1) s += __shfl_xor(s, o, 32);
  out[b * KTOP + lane] = e / s;
}

extern "C" void kernel_launch(void* const* d_in, const int* in_sizes, int n_in,
                              void* d_out, int out_size, void* d_ws, size_t ws_size,
                              hipStream_t stream) {
  (void)in_sizes; (void)n_in; (void)out_size; (void)ws_size;
  const float* patient    = (const float*)d_in[0];
  const float* treatment  = (const float*)d_in[1];
  const float* confound   = (const float*)d_in[2];
  const float* corpus     = (const float*)d_in[3];
  const float* W_p1 = (const float*)d_in[4];  const float* b_p1 = (const float*)d_in[5];
  const float* W_p2 = (const float*)d_in[6];  const float* b_p2 = (const float*)d_in[7];
  const float* W_t  = (const float*)d_in[8];  const float* b_t  = (const float*)d_in[9];
  const float* W_c  = (const float*)d_in[10]; const float* b_c  = (const float*)d_in[11];
  const float* W_r  = (const float*)d_in[12]; const float* b_r  = (const float*)d_in[13];
  const float* W_o1 = (const float*)d_in[14]; const float* b_o1 = (const float*)d_in[15];
  const float* W_o2 = (const float*)d_in[16]; const float* b_o2 = (const float*)d_in[17];
  const float* W_o3 = (const float*)d_in[18]; const float* b_o3 = (const float*)d_in[19];
  const float* W_a1 = (const float*)d_in[20]; const float* b_a1 = (const float*)d_in[21];
  const float* W_a2 = (const float*)d_in[22]; const float* b_a2 = (const float*)d_in[23];

  float* out = (float*)d_out;
  float* out_outcome = out;                      // [1024,8]
  float* out_scores  = out + 8192;               // [1024,32]
  float* out_idx     = out + 40960;              // [1024,32] (float-cast indices)
  float* out_attr    = out + 73728;              // [1024,32]
  float* out_emb     = out + 106496;             // [1024,128]

  // workspace carve-out
  char* w = (char*)d_ws;
  size_t o = 0;
  auto alloc = [&](size_t bytes) -> void* {
    o = (o + 255) & ~(size_t)255;
    void* p = w + o; o += bytes; return p;
  };
  half_t* corpusH = (half_t*)alloc((size_t)NCORP * EDIM * 2);
  half_t* queryH  = (half_t*)alloc((size_t)BDIM * EDIM * 2);
  float*  hbuf    = (float*) alloc((size_t)BDIM * HDIM * 4);
  float*  embRaw  = (float*) alloc((size_t)BDIM * EDIM * 4);
  float*  candV   = (float*) alloc((size_t)BDIM * NCAND * 4);
  int*    candI   = (int*)   alloc((size_t)BDIM * NCAND * 4);
  int*    topkI   = (int*)   alloc((size_t)BDIM * KTOP * 4);
  half_t* flatH   = (half_t*)alloc((size_t)BDIM * KTOP * EDIM * 2);
  half_t* WrtH    = (half_t*)alloc((size_t)256 * 4096 * 2);
  float*  combined= (float*) alloc((size_t)BDIM * 768 * 4);
  float*  o1buf   = (float*) alloc((size_t)BDIM * HDIM * 4);
  float*  o2buf   = (float*) alloc((size_t)BDIM * 128 * 4);
  float*  ainbuf  = (float*) alloc((size_t)BDIM * 264 * 4);
  float*  a1buf   = (float*) alloc((size_t)BDIM * HDIM * 4);
  float*  logits  = (float*) alloc((size_t)BDIM * KTOP * 4);

  // 1) corpus normalize -> fp16
  k_corpus_norm<<<(NCORP + 7) / 8, 256, 0, stream>>>(corpus, corpusH);
  // 2) patient encoder
  k_linear<<<BDIM, 256, 0, stream>>>(patient, W_p1, b_p1, hbuf, 80, HDIM, HDIM, 0, 1);
  k_linear<<<BDIM, 256, 0, stream>>>(hbuf, W_p2, b_p2, embRaw, HDIM, EDIM, EDIM, 0, 0);
  k_norm_query<<<BDIM, 128, 0, stream>>>(embRaw, out_emb, queryH);
  // 3) similarity (WMMA) + fused chunk top-k
  k_sim_topk<<<dim3(BDIM / ROWS, NCHUNK), 256, ROWS * CHUNK * 4, stream>>>(queryH, corpusH, candV, candI);
  // 4) merge to final top-32
  k_merge_topk<<<BDIM, 256, 0, stream>>>(candV, candI, out_scores, out_idx, topkI);
  // 5) gather (async->LDS DMA) + W_r transpose + r_enc (WMMA)
  k_gather<<<BDIM, 256, 0, stream>>>(corpus, topkI, flatH);
  k_wr_transpose<<<4096, 256, 0, stream>>>(W_r, WrtH);
  k_renc_wmma<<<dim3(BDIM / 16, HDIM / 16), 32, 0, stream>>>(flatH, WrtH, b_r, combined);
  // 6) t_enc / c_enc into combined
  k_linear<<<BDIM, 256, 0, stream>>>(treatment, W_t, b_t, combined, 16, HDIM, 768, 0, 0);
  k_linear<<<BDIM, 256, 0, stream>>>(confound, W_c, b_c, combined, 64, HDIM, 768, 256, 0);
  // 7) outcome MLP
  k_linear<<<BDIM, 256, 0, stream>>>(combined, W_o1, b_o1, o1buf, 768, HDIM, HDIM, 0, 1);
  k_linear<<<BDIM, 256, 0, stream>>>(o1buf, W_o2, b_o2, o2buf, HDIM, 128, 128, 0, 1);
  k_linear<<<BDIM, 256, 0, stream>>>(o2buf, W_o3, b_o3, out_outcome, 128, ODIM, ODIM, 0, 0);
  // 8) attribution network
  k_build_ain<<<BDIM, 256, 0, stream>>>(combined, out_outcome, ainbuf);
  k_linear<<<BDIM, 256, 0, stream>>>(ainbuf, W_a1, b_a1, a1buf, 264, HDIM, HDIM, 0, 1);
  k_linear<<<BDIM, 256, 0, stream>>>(a1buf, W_a2, b_a2, logits, HDIM, KTOP, KTOP, 0, 0);
  k_softmax32<<<BDIM / 8, 256, 0, stream>>>(logits, out_attr);
}
